// GNO_26568667693798
// MI455X (gfx1250) — compile-verified, matrix operand order checked by reading, not run on hardware
//
#include <hip/hip_runtime.h>

typedef __attribute__((ext_vector_type(16))) _Float16 v16h;
typedef __attribute__((ext_vector_type(8)))  float    v8f;

#define GRID_N   128
#define N_PTS    (GRID_N * GRID_N)
#define IN_DIM   32
#define OUT_DIM  32
#define PROJ_HID 256
#define K_HID    64

// Packed B-tile ids: W0[16] | W1[8kt x 2nb] | K0[4] | K1[2kt x 4nb] | K2[2kt x 2nb]
#define TILE_W0  0
#define TILE_W1  16
#define TILE_K0  32
#define TILE_K1  36
#define TILE_K2  44
#define N_TILES  48

// Branch-free exact-ish gelu: Abramowitz-Stegun 7.1.26 erf (|err| <= 1.5e-7).
__device__ __forceinline__ float gelu_fast(float x) {
    float ax = fabsf(x * 0.70710678118654752440f);       // |x|/sqrt(2)
    float t  = __builtin_amdgcn_rcpf(fmaf(0.3275911f, ax, 1.0f));
    float p  = fmaf(t, 1.061405429f, -1.453152027f);
    p = fmaf(t, p, 1.421413741f);
    p = fmaf(t, p, -0.284496736f);
    p = fmaf(t, p, 0.254829592f);
    p = p * t;
    float e  = __builtin_amdgcn_exp2f(ax * ax * -1.4426950408889634f);
    float er = copysignf(fmaf(-p, e, 1.0f), x);          // erf(x/sqrt(2))
    return 0.5f * x * (1.0f + er);
}

// K-index mapping for 16-bit A/B operands of V_WMMA_*_16X16X32 (wave32):
// lane half h, vector element i: i<8 -> K=8h+i (V0..3); i>=8 -> K=16+8h+(i-8).
__device__ __forceinline__ int kmap(int i, int h) {
    return (i < 8) ? (8 * h + i) : (16 + 8 * h + (i - 8));
}

// Load a pre-swizzled B tile: one 32B vector load per lane.
__device__ __forceinline__ v16h load_packed_tile(const _Float16* __restrict__ pk,
                                                 int tile, int lane) {
    return *(const v16h*)(pk + ((size_t)tile * 32 + lane) * 16);
}

// Activations in LDS row-major [m][k], stride in halves (mult of 8).
// A-tile load: two contiguous 16B LDS loads per lane.
__device__ __forceinline__ v16h load_act_a(const _Float16* __restrict__ lds,
                                           int stride, int kofs, int lane) {
    int h = lane >> 4, m = lane & 15;
    const _Float16* row = lds + m * stride;
    union { v16h v; float4 q[2]; } u;
    u.q[0] = *(const float4*)(row + kofs + 8 * h);        // K = kofs+8h .. +7
    u.q[1] = *(const float4*)(row + kofs + 16 + 8 * h);   // K = kofs+16+8h ..
    return u.v;
}

// Store this lane's C/D column (col fixed, rows m = 8h+v) as f16 into [m][k].
__device__ __forceinline__ void store_act_rows(_Float16* __restrict__ lds,
                                               int stride, int col,
                                               const float* g, int lane) {
    int h = lane >> 4;
#pragma unroll
    for (int v = 0; v < 8; ++v)
        lds[(8 * h + v) * stride + col] = (_Float16)g[v];
}

// ---------------------------------------------------------------------------
// Pack kernel: swizzle all fp32 weight matrices into per-lane f16 B-tile
// layout (zero-padded) once per launch. 48 blocks x 32 threads.
// ---------------------------------------------------------------------------
__global__ void __launch_bounds__(32)
gno_pack(const float* __restrict__ W0, const float* __restrict__ W1,
         const float* __restrict__ K0, const float* __restrict__ K1,
         const float* __restrict__ K2, _Float16* __restrict__ pk)
{
    int t = blockIdx.x, lane = threadIdx.x;
    const float* W; int ldn, kofs, nofs, kmax;
    if (t < TILE_W1)      { int u = t;           W = W0; ldn = PROJ_HID; kofs = 0;           nofs = u * 16;        kmax = IN_DIM;   }
    else if (t < TILE_K0) { int u = t - TILE_W1; W = W1; ldn = OUT_DIM;  kofs = (u >> 1)*32; nofs = (u & 1) * 16;  kmax = PROJ_HID; }
    else if (t < TILE_K1) { int u = t - TILE_K0; W = K0; ldn = K_HID;    kofs = 0;           nofs = u * 16;        kmax = 4;        }
    else if (t < TILE_K2) { int u = t - TILE_K1; W = K1; ldn = K_HID;    kofs = (u >> 2)*32; nofs = (u & 3) * 16;  kmax = K_HID;    }
    else                  { int u = t - TILE_K2; W = K2; ldn = OUT_DIM;  kofs = (u >> 1)*32; nofs = (u & 1) * 16;  kmax = K_HID;    }

    int h = lane >> 4, n = lane & 15;
    v16h b;
#pragma unroll
    for (int i = 0; i < 16; ++i) {
        int k = kofs + kmap(i, h);
        b[i] = (k < kmax) ? (_Float16)W[k * ldn + nofs + n] : (_Float16)0.0f;
    }
    *(v16h*)(pk + ((size_t)t * 32 + lane) * 16) = b;
}

// ---------------------------------------------------------------------------
// Kernel 1: f = gelu(x @ W0 + b0) @ W1 + b1   (16 rows per wave, 32 WMMAs)
// ---------------------------------------------------------------------------
__global__ void __launch_bounds__(32)
gno_features(const float* __restrict__ x,    // [N_PTS][IN_DIM]
             const float* __restrict__ b0v,  // [PROJ_HID]
             const float* __restrict__ b1v,  // [OUT_DIM]
             const _Float16* __restrict__ pk,
             float* __restrict__ f_ws)       // [N_PTS][OUT_DIM]
{
    constexpr int LDB = PROJ_HID + 8;        // padded stride (halves)
    __shared__ _Float16 h1[16 * LDB];

    int lane = threadIdx.x;
    int h = lane >> 4, n = lane & 15;
    int r0 = blockIdx.x * 16;

    // A-tile of x: row = r0 + n; per-lane K runs are contiguous -> float4 loads
    const float* xr = x + (size_t)(r0 + n) * IN_DIM;
    float4 p0 = *(const float4*)(xr + 8 * h);
    float4 p1 = *(const float4*)(xr + 8 * h + 4);
    float4 p2 = *(const float4*)(xr + 16 + 8 * h);
    float4 p3 = *(const float4*)(xr + 20 + 8 * h);
    float tmp[16] = {p0.x, p0.y, p0.z, p0.w, p1.x, p1.y, p1.z, p1.w,
                     p2.x, p2.y, p2.z, p2.w, p3.x, p3.y, p3.z, p3.w};
    v16h a;
#pragma unroll
    for (int i = 0; i < 16; ++i) a[i] = (_Float16)tmp[i];

    // Layer 1: 16 N-tiles, K=32
#pragma unroll
    for (int nb = 0; nb < 16; ++nb) {
        v16h b = load_packed_tile(pk, TILE_W0 + nb, lane);
        float bias = b0v[nb * 16 + n];
        v8f c;
#pragma unroll
        for (int v = 0; v < 8; ++v) c[v] = bias;
        c = __builtin_amdgcn_wmma_f32_16x16x32_f16(false, a, false, b,
                                                   (short)0, c, false, false);
        float g[8];
#pragma unroll
        for (int v = 0; v < 8; ++v) g[v] = gelu_fast(c[v]);
        store_act_rows(h1, LDB, nb * 16 + n, g, lane);
    }
    __syncthreads();

    // Layer 2: K=256 (8 K-blocks), both N-tiles accumulated per A load
    v8f c0, c1;
    {
        float bb0 = b1v[n], bb1 = b1v[16 + n];
#pragma unroll
        for (int v = 0; v < 8; ++v) { c0[v] = bb0; c1[v] = bb1; }
    }
#pragma unroll
    for (int kt = 0; kt < 8; ++kt) {
        v16h a2 = load_act_a(h1, LDB, kt * 32, lane);
        v16h bA = load_packed_tile(pk, TILE_W1 + kt * 2 + 0, lane);
        v16h bB = load_packed_tile(pk, TILE_W1 + kt * 2 + 1, lane);
        c0 = __builtin_amdgcn_wmma_f32_16x16x32_f16(false, a2, false, bA,
                                                    (short)0, c0, false, false);
        c1 = __builtin_amdgcn_wmma_f32_16x16x32_f16(false, a2, false, bB,
                                                    (short)0, c1, false, false);
    }
#pragma unroll
    for (int v = 0; v < 8; ++v) {
        int row = r0 + 8 * h + v;
        f_ws[(size_t)row * OUT_DIM + n]      = c0[v];
        f_ws[(size_t)row * OUT_DIM + 16 + n] = c1[v];
    }
}

// ---------------------------------------------------------------------------
// Kernel 2: per-edge MLP (4->64->64->32) * f[nb], segment mean.
// One wave per output point; B tiles + biases live in registers.
// ---------------------------------------------------------------------------
__global__ void __launch_bounds__(32)
gno_edges(const float* __restrict__ in_grid,   // [N_PTS][2]
          const float* __restrict__ out_grid,  // [N_PTS][2]
          const float* __restrict__ kb0, const float* __restrict__ kb1,
          const float* __restrict__ kb2,
          const int* __restrict__ nb_index,
          const int* __restrict__ seg_ids,
          const int* __restrict__ counts,
          const float* __restrict__ f_ws,      // [N_PTS][OUT_DIM]
          const _Float16* __restrict__ pk,
          float* __restrict__ out,             // [N_PTS][OUT_DIM]
          int E)
{
    constexpr int LDB = K_HID + 8;             // padded stride (halves)
    __shared__ _Float16 h1[16 * LDB];
    __shared__ _Float16 h2[16 * LDB];
    __shared__ int nbs[16];

    int lane = threadIdx.x;
    int h = lane >> 4, n = lane & 15;
    int q = blockIdx.x;

    // Hoist all weight tiles into registers (16 x v16h = 128 VGPRs).
    v16h bK0[4], bK1[8], bK2[4];
#pragma unroll
    for (int i = 0; i < 4; ++i) bK0[i] = load_packed_tile(pk, TILE_K0 + i, lane);
#pragma unroll
    for (int i = 0; i < 8; ++i) bK1[i] = load_packed_tile(pk, TILE_K1 + i, lane);
#pragma unroll
    for (int i = 0; i < 4; ++i) bK2[i] = load_packed_tile(pk, TILE_K2 + i, lane);

    float bias0[4], bias1[4], bias2[2];
#pragma unroll
    for (int i = 0; i < 4; ++i) { bias0[i] = kb0[i * 16 + n]; bias1[i] = kb1[i * 16 + n]; }
#pragma unroll
    for (int i = 0; i < 2; ++i) bias2[i] = kb2[i * 16 + n];

    // lower_bound(seg_ids, q) — uniform across the wave
    int lo = 0, hi = E;
    while (lo < hi) {
        int mid = (lo + hi) >> 1;
        if (seg_ids[mid] < q) lo = mid + 1; else hi = mid;
    }
    int start = lo;
    int cnt = counts[q];

    float sx = out_grid[q * 2 + 0];
    float sy = out_grid[q * 2 + 1];

    float acc0 = 0.0f, acc1 = 0.0f;

    for (int t0 = 0; t0 < cnt; t0 += 16) {
        int rem = cnt - t0;
        if (lane < 16) {
            int e = start + t0 + ((lane < rem) ? lane : 0);   // clamp pad rows
            nbs[lane] = nb_index[e];
        }
        __syncthreads();

        // ---- Layer 1: agg = [rep.x, rep.y, slf.x, slf.y] (K=4 padded to 32)
        v16h a1 = {};
        if (h == 0) {
            const float* rp = &in_grid[(size_t)nbs[n] * 2];
            a1[0] = (_Float16)rp[0];
            a1[1] = (_Float16)rp[1];
            a1[2] = (_Float16)sx;
            a1[3] = (_Float16)sy;
        }
#pragma unroll
        for (int nb = 0; nb < 4; ++nb) {
            v8f c;
#pragma unroll
            for (int v = 0; v < 8; ++v) c[v] = bias0[nb];
            c = __builtin_amdgcn_wmma_f32_16x16x32_f16(false, a1, false, bK0[nb],
                                                       (short)0, c, false, false);
            float g[8];
#pragma unroll
            for (int v = 0; v < 8; ++v) g[v] = gelu_fast(c[v]);
            store_act_rows(h1, LDB, nb * 16 + n, g, lane);
        }
        __syncthreads();

        // ---- Layer 2: 64 -> 64
        v16h a20 = load_act_a(h1, LDB, 0,  lane);
        v16h a21 = load_act_a(h1, LDB, 32, lane);
#pragma unroll
        for (int nb = 0; nb < 4; ++nb) {
            v8f c;
#pragma unroll
            for (int v = 0; v < 8; ++v) c[v] = bias1[nb];
            c = __builtin_amdgcn_wmma_f32_16x16x32_f16(false, a20, false, bK1[nb],
                                                       (short)0, c, false, false);
            c = __builtin_amdgcn_wmma_f32_16x16x32_f16(false, a21, false, bK1[4 + nb],
                                                       (short)0, c, false, false);
            float g[8];
#pragma unroll
            for (int v = 0; v < 8; ++v) g[v] = gelu_fast(c[v]);
            store_act_rows(h2, LDB, nb * 16 + n, g, lane);
        }
        __syncthreads();

        // ---- Layer 3: 64 -> 32, scale by gathered f, reduce rows
        v16h a30 = load_act_a(h2, LDB, 0,  lane);
        v16h a31 = load_act_a(h2, LDB, 32, lane);
#pragma unroll
        for (int nb = 0; nb < 2; ++nb) {
            v8f c;
#pragma unroll
            for (int v = 0; v < 8; ++v) c[v] = bias2[nb];
            c = __builtin_amdgcn_wmma_f32_16x16x32_f16(false, a30, false, bK2[nb],
                                                       (short)0, c, false, false);
            c = __builtin_amdgcn_wmma_f32_16x16x32_f16(false, a31, false, bK2[2 + nb],
                                                       (short)0, c, false, false);
            float part = 0.0f;
#pragma unroll
            for (int v = 0; v < 8; ++v) {
                int m = 8 * h + v;
                float fv = (t0 + m < cnt)
                         ? f_ws[(size_t)nbs[m] * OUT_DIM + nb * 16 + n]
                         : 0.0f;                    // mask padded rows
                part = fmaf(c[v], fv, part);
            }
            if (nb == 0) acc0 += part; else acc1 += part;
        }
        __syncthreads();   // protect nbs/h1/h2 for next tile
    }

    // Combine wave halves (rows 0-7 vs 8-15 of each tile)
    acc0 += __shfl_xor(acc0, 16, 32);
    acc1 += __shfl_xor(acc1, 16, 32);

    float denom = (float)((cnt > 0) ? cnt : 1);
    int col   = (lane < 16) ? n : (16 + n);
    float val = ((lane < 16) ? acc0 : acc1) / denom;
    out[(size_t)q * OUT_DIM + col] = val;
}

// ---------------------------------------------------------------------------
extern "C" void kernel_launch(void* const* d_in, const int* in_sizes, int n_in,
                              void* d_out, int out_size, void* d_ws, size_t ws_size,
                              hipStream_t stream) {
    const float* inp      = (const float*)d_in[0];
    const float* in_grid  = (const float*)d_in[1];
    const float* out_grid = (const float*)d_in[2];
    const float* W0       = (const float*)d_in[3];
    const float* b0       = (const float*)d_in[4];
    const float* W1       = (const float*)d_in[5];
    const float* b1       = (const float*)d_in[6];
    const float* K0       = (const float*)d_in[7];
    const float* kb0      = (const float*)d_in[8];
    const float* K1       = (const float*)d_in[9];
    const float* kb1      = (const float*)d_in[10];
    const float* K2       = (const float*)d_in[11];
    const float* kb2      = (const float*)d_in[12];
    const int*   nb_index = (const int*)d_in[13];
    const int*   seg_ids  = (const int*)d_in[14];
    const int*   counts   = (const int*)d_in[15];
    int E = in_sizes[13];

    float*     f_ws = (float*)d_ws;                         // 2 MB
    _Float16*  pk   = (_Float16*)((char*)d_ws +
                                  (size_t)N_PTS * OUT_DIM * sizeof(float));
    float*     out  = (float*)d_out;

    gno_pack<<<N_TILES, 32, 0, stream>>>(W0, W1, K0, K1, K2, pk);
    gno_features<<<N_PTS / 16, 32, 0, stream>>>(inp, b0, b1, pk, f_ws);
    gno_edges<<<N_PTS, 32, 0, stream>>>(in_grid, out_grid, kb0, kb1, kb2,
                                        nb_index, seg_ids, counts,
                                        f_ws, pk, out, E);
}
